// ContourFormerDecoderLayer_75539884802136
// MI455X (gfx1250) — matrix operand order repaired and checked by reading, not compiled, not run on hardware
//
#include <hip/hip_runtime.h>
#include <stdint.h>
#include <stddef.h>

// ---------------------------------------------------------------------------
// ContourFormer decoder layer for MI455X (gfx1250, wave32, WMMA).
// GEMMs run on v_wmma_f32_16x16x32_bf16 (fp32 accum). Attention is a
// flash-style wave kernel with async global->LDS staging; LN uses wave32
// shfl reductions; deformable sampling is a wave-per-(b,lq,head) gather.
// ---------------------------------------------------------------------------

typedef __attribute__((ext_vector_type(16))) __bf16 v16bf;
typedef __attribute__((ext_vector_type(8)))  __bf16 v8bf;
typedef __attribute__((ext_vector_type(8)))  float  v8f;

#define BB    4
#define NQ    100
#define NC    36
#define CC    256
#define HN    8
#define PP    4
#define DFF   2048
#define HH    128
#define WW    128
#define LQ    (NQ * NC)          // 3600
#define ROWS  (BB * LQ)          // 14400
#define MI    (BB * NQ)          // 400
#define LN_EPSF 1e-5f

__device__ __forceinline__ int imin(int a, int b) { return a < b ? a : b; }

__device__ __forceinline__ __bf16 f2bf(float f) {
  union { float f; uint32_t u; } v; v.f = f;
  uint32_t r = v.u + 0x7FFFu + ((v.u >> 16) & 1u);   // round-to-nearest-even
  uint16_t h = (uint16_t)(r >> 16);
  return __builtin_bit_cast(__bf16, h);
}

__device__ __forceinline__ float wsum(float v) {
#pragma unroll
  for (int o = 16; o > 0; o >>= 1) v += __shfl_xor(v, o, 32);
  return v;
}

// A fragment, 16-bit A 16x32 ISA layout: lane<16 holds K {koff..koff+7, koff+16..koff+23}
__device__ __forceinline__ v16bf ldfragA(const __bf16* p) {
  v8bf a0 = *(const v8bf*)(p);
  v8bf a1 = *(const v8bf*)(p + 16);
  return __builtin_shufflevector(a0, a1, 0,1,2,3,4,5,6,7,8,9,10,11,12,13,14,15);
}
// B fragment, same K order as A: lane<16 holds K {koff..koff+7, koff+16..koff+23}
__device__ __forceinline__ v16bf ldfragB(const __bf16* p) {
  v8bf b0 = *(const v8bf*)(p);
  v8bf b1 = *(const v8bf*)(p + 16);
  return __builtin_shufflevector(b0, b1, 0,1,2,3,4,5,6,7,8,9,10,11,12,13,14,15);
}

// D[M,N] = A[M,K](bf16) @ W[N,K]^T(bf16) + bias, optional relu; fp32/bf16 outs.
__global__ void __launch_bounds__(256)
k_gemm_bf16(const __bf16* __restrict__ A, const __bf16* __restrict__ W,
            const float* __restrict__ bias, float* __restrict__ outF,
            __bf16* __restrict__ outB, int M, int N, int K, int relu)
{
  const int lane = threadIdx.x & 31;
  const int wave = threadIdx.x >> 5;        // 0..7
  const int wm = wave >> 1, wn = wave & 1;  // 4x2 wave grid
  const int m0 = blockIdx.y * 128 + wm * 32;
  const int n0 = blockIdx.x * 64 + wn * 32;
  const int l16 = lane & 15;
  const int lhi = lane >> 4;
  const int koffA = lhi * 8;
  const int koffB = lhi * 8;
  const size_t sk = (size_t)K;

  const __bf16* pA0 = A + (size_t)imin(m0 + l16,      M - 1) * sk + koffA;
  const __bf16* pA1 = A + (size_t)imin(m0 + 16 + l16, M - 1) * sk + koffA;
  const __bf16* pB0 = W + (size_t)imin(n0 + l16,      N - 1) * sk + koffB;
  const __bf16* pB1 = W + (size_t)imin(n0 + 16 + l16, N - 1) * sk + koffB;

  v8f acc[2][2] = {};
  for (int k = 0; k < K; k += 32) {
    if (k + 128 < K) {
      __builtin_prefetch(pA0 + k + 128, 0, 3);   // global_prefetch_b8, near-temporal
      __builtin_prefetch(pB0 + k + 128, 0, 3);
    }
    v16bf a0 = ldfragA(pA0 + k);
    v16bf a1 = ldfragA(pA1 + k);
    v16bf b0 = ldfragB(pB0 + k);
    v16bf b1 = ldfragB(pB1 + k);
    acc[0][0] = __builtin_amdgcn_wmma_f32_16x16x32_bf16(false, a0, false, b0, (short)0, acc[0][0], false, false);
    acc[0][1] = __builtin_amdgcn_wmma_f32_16x16x32_bf16(false, a0, false, b1, (short)0, acc[0][1], false, false);
    acc[1][0] = __builtin_amdgcn_wmma_f32_16x16x32_bf16(false, a1, false, b0, (short)0, acc[1][0], false, false);
    acc[1][1] = __builtin_amdgcn_wmma_f32_16x16x32_bf16(false, a1, false, b1, (short)0, acc[1][1], false, false);
  }

#pragma unroll
  for (int i = 0; i < 2; ++i) {
#pragma unroll
    for (int j = 0; j < 2; ++j) {
      int n = n0 + j * 16 + l16;
      if (n >= N) continue;
      float bv = bias ? bias[n] : 0.0f;
#pragma unroll
      for (int r = 0; r < 8; ++r) {
        int m = m0 + i * 16 + lhi * 8 + r;     // 16x16 f32 C/D layout
        if (m >= M) continue;
        float v = acc[i][j][r] + bv;
        if (relu) v = fmaxf(v, 0.0f);
        size_t idx = (size_t)m * N + n;
        if (outF) outF[idx] = v;
        if (outB) outB[idx] = f2bf(v);
      }
    }
  }
}

__global__ void k_cvt_bf16(const float* __restrict__ x, __bf16* __restrict__ y, size_t n) {
  size_t st = (size_t)gridDim.x * blockDim.x;
  for (size_t i = (size_t)blockIdx.x * blockDim.x + threadIdx.x; i < n; i += st)
    y[i] = f2bf(x[i]);
}

__global__ void k_copy_f32(const float* __restrict__ x, float* __restrict__ y, size_t n) {
  size_t st = (size_t)gridDim.x * blockDim.x;
  for (size_t i = (size_t)blockIdx.x * blockDim.x + threadIdx.x; i < n; i += st)
    y[i] = x[i];
}

// [B,Nq,Nc,C] f32 -> [B,Nc,Nq,C] bf16
__global__ void k_sc_transpose(const float* __restrict__ x, __bf16* __restrict__ y) {
  const size_t n = (size_t)BB * NQ * NC * CC;
  size_t st = (size_t)gridDim.x * blockDim.x;
  for (size_t i = (size_t)blockIdx.x * blockDim.x + threadIdx.x; i < n; i += st) {
    int c = (int)(i & (CC - 1));
    size_t r = i >> 8;
    int nq = (int)(r % NQ);
    size_t r2 = r / NQ;
    int nc = (int)(r2 % 36);
    int b  = (int)(r2 / 36);
    y[i] = f2bf(x[(((size_t)(b * NQ + nq)) * NC + nc) * CC + c]);
  }
}

// Flash attention over Nq=100 tokens, hd=32. qkv: [nBatch*NQ, 768]; one block per (batch,head).
// K/V staged into LDS with CDNA5 async global->LDS DMA (ASYNCcnt-tracked).
__global__ void __launch_bounds__(128)
k_attn(const float* __restrict__ qkv, __bf16* __restrict__ outB)
{
  __shared__ float sK[NQ * 32];
  __shared__ float sV[NQ * 32];
  const int batch = blockIdx.x >> 3;
  const int h = blockIdx.x & 7;
  const int tid = threadIdx.x;
  const float* base = qkv + (size_t)batch * NQ * 768;

  for (int i = tid; i < NQ * 32; i += 128) {
    int r = i >> 5, d = i & 31;
    const float* gk = base + (size_t)r * 768 + 256 + h * 32 + d;
    const float* gv = base + (size_t)r * 768 + 512 + h * 32 + d;
    // Generic LDS pointer: low 32 bits are the LDS byte address (aperture rule).
    unsigned lk = (unsigned)(uintptr_t)(&sK[i]);
    unsigned lv = (unsigned)(uintptr_t)(&sV[i]);
    asm volatile("global_load_async_to_lds_b32 %0, %1, off"
                 :: "v"(lk), "v"(gk) : "memory");
    asm volatile("global_load_async_to_lds_b32 %0, %1, off"
                 :: "v"(lv), "v"(gv) : "memory");
  }
  asm volatile("s_wait_asynccnt 0" ::: "memory");
  __syncthreads();
  if (tid >= NQ) return;

  float q[32];
#pragma unroll
  for (int d = 0; d < 32; ++d) q[d] = base[(size_t)tid * 768 + h * 32 + d];
  const float scale = 0.17677669529663687f;  // 1/sqrt(32)

  float m = -3.0e38f, s = 0.0f;
  float acc[32];
#pragma unroll
  for (int d = 0; d < 32; ++d) acc[d] = 0.0f;

  for (int j = 0; j < NQ; ++j) {
    float sc = 0.0f;
#pragma unroll
    for (int d = 0; d < 32; ++d) sc += q[d] * sK[j * 32 + d];
    sc *= scale;
    float mn = fmaxf(m, sc);
    float corr = __expf(m - mn);
    float p = __expf(sc - mn);
    s = s * corr + p;
#pragma unroll
    for (int d = 0; d < 32; ++d) acc[d] = acc[d] * corr + p * sV[j * 32 + d];
    m = mn;
  }
  float inv = 1.0f / s;
  __bf16* o = outB + ((size_t)batch * NQ + tid) * CC + h * 32;
#pragma unroll
  for (int d = 0; d < 32; ++d) o[d] = f2bf(acc[d] * inv);
}

// tgt = LN(sub + 2*(sc_proj^T + inst_out)); wave per row, lane holds 8 channels.
__global__ void __launch_bounds__(256)
k_qsa_ln(const float* __restrict__ sub, const float* __restrict__ scp,
         const float* __restrict__ inst, const float* __restrict__ g,
         const float* __restrict__ be, float* __restrict__ outF,
         __bf16* __restrict__ outB)
{
  const int wave = threadIdx.x >> 5, lane = threadIdx.x & 31;
  const int row = blockIdx.x * 8 + wave;
  if (row >= ROWS) return;
  const int b = row / LQ;
  const int rem = row % LQ;
  const int nq = rem / NC;
  const int nc = rem % NC;
  const float* ps = sub + (size_t)row * CC;
  const float* pc = scp + ((size_t)((b * NC + nc) * NQ + nq)) * CC;
  const float* pi = inst + (size_t)(b * NQ + nq) * CC;
  const int c0 = lane * 8;

  float x[8];
  float lsum = 0.0f;
#pragma unroll
  for (int i = 0; i < 8; ++i) { x[i] = ps[c0 + i] + 2.0f * (pc[c0 + i] + pi[c0 + i]); lsum += x[i]; }
  float mean = wsum(lsum) * (1.0f / CC);
  float lvar = 0.0f;
#pragma unroll
  for (int i = 0; i < 8; ++i) { float d = x[i] - mean; lvar += d * d; }
  float rstd = rsqrtf(wsum(lvar) * (1.0f / CC) + LN_EPSF);
#pragma unroll
  for (int i = 0; i < 8; ++i) {
    float y = (x[i] - mean) * rstd * g[c0 + i] + be[c0 + i];
    outF[(size_t)row * CC + c0 + i] = y;
    outB[(size_t)row * CC + c0 + i] = f2bf(y);
  }
}

// out = LN(a + b2); bf16 output optional; safe in-place (row staged in regs).
__global__ void __launch_bounds__(256)
k_ln_add(const float* __restrict__ a, const float* __restrict__ b2,
         const float* __restrict__ g, const float* __restrict__ be,
         float* __restrict__ outF, __bf16* __restrict__ outB)
{
  const int wave = threadIdx.x >> 5, lane = threadIdx.x & 31;
  const int row = blockIdx.x * 8 + wave;
  if (row >= ROWS) return;
  const int c0 = lane * 8;
  float x[8];
  float lsum = 0.0f;
#pragma unroll
  for (int i = 0; i < 8; ++i) { x[i] = a[(size_t)row * CC + c0 + i] + b2[(size_t)row * CC + c0 + i]; lsum += x[i]; }
  float mean = wsum(lsum) * (1.0f / CC);
  float lvar = 0.0f;
#pragma unroll
  for (int i = 0; i < 8; ++i) { float d = x[i] - mean; lvar += d * d; }
  float rstd = rsqrtf(wsum(lvar) * (1.0f / CC) + LN_EPSF);
#pragma unroll
  for (int i = 0; i < 8; ++i) {
    float y = (x[i] - mean) * rstd * g[c0 + i] + be[c0 + i];
    outF[(size_t)row * CC + c0 + i] = y;
    if (outB) outB[(size_t)row * CC + c0 + i] = f2bf(y);
  }
}

// Deformable attention sampling: one wave per (b, lq, head); lane = channel.
__global__ void __launch_bounds__(256)
k_deform(const float* __restrict__ mem, const float* __restrict__ refb,
         const float* __restrict__ offs, const float* __restrict__ awr,
         __bf16* __restrict__ outB)
{
  const int wave = threadIdx.x >> 5, lane = threadIdx.x & 31;
  const int wid = blockIdx.x * 8 + wave;
  if (wid >= ROWS * HN) return;
  const int h = wid & 7;
  const int t = wid >> 3;          // b*LQ + lq
  const int b = t / LQ;

  const float* pa = awr + ((size_t)t * HN + h) * PP;
  float a0 = pa[0], a1 = pa[1], a2 = pa[2], a3 = pa[3];
  float mx = fmaxf(fmaxf(a0, a1), fmaxf(a2, a3));
  float e0 = __expf(a0 - mx), e1 = __expf(a1 - mx), e2 = __expf(a2 - mx), e3 = __expf(a3 - mx);
  float inv = 1.0f / (e0 + e1 + e2 + e3);
  float wp[PP] = {e0 * inv, e1 * inv, e2 * inv, e3 * inv};

  const float* pr = refb + (size_t)t * 4;
  float cx = pr[0], cy = pr[1], bw = pr[2], bh = pr[3];
  const float* po = offs + ((size_t)t * HN + h) * PP * 2;
  const float* pm = mem + (size_t)b * (HH * WW * CC) + h * 32 + lane;

  float acc = 0.0f;
#pragma unroll
  for (int p = 0; p < PP; ++p) {
    float lx = cx + po[p * 2 + 0] * bw * 0.5f;
    float ly = cy + po[p * 2 + 1] * bh * 0.5f;
    float xx = lx * (float)WW - 0.5f;   // ((2l-1)+1)*W/2 - 0.5
    float yy = ly * (float)HH - 0.5f;
    float xf = floorf(xx), yf = floorf(yy);
    int x0 = (int)xf, y0 = (int)yf;
    float wx1 = xx - xf, wy1 = yy - yf;
    float wx0 = 1.0f - wx1, wy0 = 1.0f - wy1;
    float v = 0.0f;
    if ((unsigned)x0       < (unsigned)WW && (unsigned)y0       < (unsigned)HH)
      v += wy0 * wx0 * pm[((size_t)y0 * WW + x0) * CC];
    if ((unsigned)(x0 + 1) < (unsigned)WW && (unsigned)y0       < (unsigned)HH)
      v += wy0 * wx1 * pm[((size_t)y0 * WW + x0 + 1) * CC];
    if ((unsigned)x0       < (unsigned)WW && (unsigned)(y0 + 1) < (unsigned)HH)
      v += wy1 * wx0 * pm[((size_t)(y0 + 1) * WW + x0) * CC];
    if ((unsigned)(x0 + 1) < (unsigned)WW && (unsigned)(y0 + 1) < (unsigned)HH)
      v += wy1 * wx1 * pm[((size_t)(y0 + 1) * WW + x0 + 1) * CC];
    acc += wp[p] * v;
  }
  outB[(size_t)t * CC + h * 32 + lane] = f2bf(acc);
}

// ---------------------------------------------------------------------------
extern "C" void kernel_launch(void* const* d_in, const int* in_sizes, int n_in,
                              void* d_out, int out_size, void* d_ws, size_t ws_size,
                              hipStream_t stream)
{
  (void)in_sizes; (void)n_in; (void)out_size; (void)ws_size;
  const float* inst_query = (const float*)d_in[0];
  const float* sub_q      = (const float*)d_in[1];
  const float* mem        = (const float*)d_in[2];
  const float* refb       = (const float*)d_in[3];
  const float* inst_in_w  = (const float*)d_in[4];
  const float* inst_in_b  = (const float*)d_in[5];
  const float* inst_out_w = (const float*)d_in[6];
  const float* inst_out_b = (const float*)d_in[7];
  const float* sc_in_w    = (const float*)d_in[8];
  const float* sc_in_b    = (const float*)d_in[9];
  const float* sc_out_w   = (const float*)d_in[10];
  const float* sc_out_b   = (const float*)d_in[11];
  const float* off_w      = (const float*)d_in[12];
  const float* off_b      = (const float*)d_in[13];
  const float* aw_w       = (const float*)d_in[14];
  const float* aw_b       = (const float*)d_in[15];
  const float* cross_w    = (const float*)d_in[16];
  const float* cross_b    = (const float*)d_in[17];
  const float* lin1_w     = (const float*)d_in[18];
  const float* lin1_b     = (const float*)d_in[19];
  const float* lin2_w     = (const float*)d_in[20];
  const float* lin2_b     = (const float*)d_in[21];
  const float* n1_g = (const float*)d_in[22]; const float* n1_b = (const float*)d_in[23];
  const float* n2_g = (const float*)d_in[24]; const float* n2_b = (const float*)d_in[25];
  const float* n3_g = (const float*)d_in[26]; const float* n3_b = (const float*)d_in[27];

  char* ws = (char*)d_ws;
  size_t off = 0;
  auto alloc = [&](size_t bytes) -> void* {
    void* p = ws + off;
    off = (off + bytes + 255) & ~(size_t)255;
    return p;
  };

  __bf16* w_inst_in  = (__bf16*)alloc((size_t)768 * CC * 2);
  __bf16* w_inst_out = (__bf16*)alloc((size_t)CC * CC * 2);
  __bf16* w_sc_in    = (__bf16*)alloc((size_t)768 * CC * 2);
  __bf16* w_sc_out   = (__bf16*)alloc((size_t)CC * CC * 2);
  __bf16* w_off      = (__bf16*)alloc((size_t)64 * CC * 2);
  __bf16* w_aw       = (__bf16*)alloc((size_t)32 * CC * 2);
  __bf16* w_cross    = (__bf16*)alloc((size_t)CC * CC * 2);
  __bf16* w_lin1     = (__bf16*)alloc((size_t)DFF * CC * 2);
  __bf16* w_lin2     = (__bf16*)alloc((size_t)CC * DFF * 2);
  __bf16* iq_bf      = (__bf16*)alloc((size_t)MI * CC * 2);
  float*  qkv_inst   = (float*) alloc((size_t)MI * 768 * 4);
  __bf16* inst_attn  = (__bf16*)alloc((size_t)MI * CC * 2);
  float*  inst_out   = (float*) alloc((size_t)MI * CC * 4);
  __bf16* sc_bf      = (__bf16*)alloc((size_t)ROWS * CC * 2);
  float*  qkv_sc     = (float*) alloc((size_t)ROWS * 768 * 4);
  __bf16* sc_attn    = (__bf16*)alloc((size_t)ROWS * CC * 2);
  float*  sc_proj    = (float*) alloc((size_t)ROWS * CC * 4);
  float*  tgt        = (float*) alloc((size_t)ROWS * CC * 4);
  __bf16* tgt_bf     = (__bf16*)alloc((size_t)ROWS * CC * 2);
  float*  off_out    = (float*) alloc((size_t)ROWS * 64 * 4);
  float*  aw_out     = (float*) alloc((size_t)ROWS * 32 * 4);
  __bf16* wtd_bf     = (__bf16*)alloc((size_t)ROWS * CC * 2);
  float*  tgt2       = (float*) alloc((size_t)ROWS * CC * 4);
  __bf16* hid_bf     = (__bf16*)alloc((size_t)ROWS * DFF * 2);
  float*  ffn_out    = (float*) alloc((size_t)ROWS * CC * 4);

  auto cvt = [&](const float* x, __bf16* y, size_t n) {
    unsigned g = (unsigned)((n + 1023) / 1024);
    if (g > 2048u) g = 2048u;
    k_cvt_bf16<<<dim3(g), 256, 0, stream>>>(x, y, n);
  };
  auto gemm = [&](const __bf16* A, const __bf16* W, const float* bias,
                  float* oF, __bf16* oB, int M, int N, int K, int relu) {
    dim3 g((unsigned)((N + 63) / 64), (unsigned)((M + 127) / 128));
    k_gemm_bf16<<<g, 256, 0, stream>>>(A, W, bias, oF, oB, M, N, K, relu);
  };

  // Weight down-converts
  cvt(inst_in_w,  w_inst_in,  (size_t)768 * CC);
  cvt(inst_out_w, w_inst_out, (size_t)CC * CC);
  cvt(sc_in_w,    w_sc_in,    (size_t)768 * CC);
  cvt(sc_out_w,   w_sc_out,   (size_t)CC * CC);
  cvt(off_w,      w_off,      (size_t)64 * CC);
  cvt(aw_w,       w_aw,       (size_t)32 * CC);
  cvt(cross_w,    w_cross,    (size_t)CC * CC);
  cvt(lin1_w,     w_lin1,     (size_t)DFF * CC);
  cvt(lin2_w,     w_lin2,     (size_t)CC * DFF);

  // Instance self-attention
  cvt(inst_query, iq_bf, (size_t)MI * CC);
  gemm(iq_bf, w_inst_in, inst_in_b, qkv_inst, nullptr, MI, 768, CC, 0);
  k_attn<<<dim3(BB * HN), 128, 0, stream>>>(qkv_inst, inst_attn);
  gemm(inst_attn, w_inst_out, inst_out_b, inst_out, nullptr, MI, CC, CC, 0);

  // Sub-contour self-attention ([B,Nc] batched over Nq)
  k_sc_transpose<<<2048, 256, 0, stream>>>(sub_q, sc_bf);
  gemm(sc_bf, w_sc_in, sc_in_b, qkv_sc, nullptr, ROWS, 768, CC, 0);
  k_attn<<<dim3(BB * NC * HN), 128, 0, stream>>>(qkv_sc, sc_attn);
  gemm(sc_attn, w_sc_out, sc_out_b, sc_proj, nullptr, ROWS, CC, CC, 0);

  // tgt = LN(sub + 2*(sc_out + inst_out))
  k_qsa_ln<<<dim3(ROWS / 8), 256, 0, stream>>>(sub_q, sc_proj, inst_out, n1_g, n1_b, tgt, tgt_bf);

  // Deformable cross-attention
  gemm(tgt_bf, w_off, off_b, off_out, nullptr, ROWS, 64, CC, 0);
  gemm(tgt_bf, w_aw,  aw_b,  aw_out,  nullptr, ROWS, 32, CC, 0);
  k_deform<<<dim3(ROWS * HN / 8), 256, 0, stream>>>(mem, refb, off_out, aw_out, wtd_bf);
  gemm(wtd_bf, w_cross, cross_b, tgt2, nullptr, ROWS, CC, CC, 0);
  k_ln_add<<<dim3(ROWS / 8), 256, 0, stream>>>(tgt, tgt2, n2_g, n2_b, tgt, tgt_bf);

  // FFN + final LN straight into d_out
  gemm(tgt_bf, w_lin1, lin1_b, nullptr, hid_bf, ROWS, DFF, CC, 1);
  gemm(hid_bf, w_lin2, lin2_b, ffn_out, nullptr, ROWS, CC, DFF, 0);
  k_ln_add<<<dim3(ROWS / 8), 256, 0, stream>>>(tgt, ffn_out, n3_g, n3_b,
                                               (float*)d_out + (size_t)MI * CC, nullptr);

  // First tuple element: inst_query passthrough
  k_copy_f32<<<dim3(512), 256, 0, stream>>>(inst_query, (float*)d_out, (size_t)MI * CC);
}
